// CoMPILE_52905407152970
// MI455X (gfx1250) — matrix operand order repaired and checked by reading, not compiled, not run on hardware
//
#include <hip/hip_runtime.h>
#include <cstddef>

// ---------------------------------------------------------------------------
// CoMPILE forward for MI455X (gfx1250): bf16 WMMA GEMMs with fused
// gather/concat A-loaders + fused epilogues, atomic segment-sum, persistent
// bidirectional GRU kernel.
// ---------------------------------------------------------------------------

typedef __bf16 bf16_t;
typedef bf16_t v16bf __attribute__((ext_vector_type(16)));
typedef float  v8f   __attribute__((ext_vector_type(8)));

// Problem constants (fixed by the reference)
constexpr int B_    = 128;     // graphs
constexpr int NN_   = 100;     // nodes / graph
constexpr int EE_   = 800;     // edges / graph
constexpr int TN_   = B_ * NN_;   // 12800
constexpr int TE_   = B_ * EE_;   // 102400
constexpr int H_    = 256;
constexpr int NEMB_ = 128;
constexpr int REMB_ = 256;

#define DEV_ __device__ __forceinline__

DEV_ float sigf(float x) { return 1.0f / (1.0f + __expf(-x)); }

// ------------------------------- GEMM -------------------------------------
// C[M,N] = act( A[M,K] @ W^T + bias + res ),  W row-major [N,K] (fp32, cast
// to bf16 on LDS staging). A sources are virtual (gather/concat modes).

enum {
  AM_PLAIN = 0,   // a0 fp32, leading dim lda
  AM_ADD2,        // a0 + a1, both fp32 [*,H]
  AM_GATHER,      // a0 fp32 [*,H] gathered by idx[m]
  AM_CONCATQ,     // k<H: a0 bf16 [*,H];  k>=H: a1 fp32 [B,H] indexed by m/epg
  AM_CONCAT2,     // a0 | a1, fp32 [*,H] each
  AM_CONCAT3,     // a0 | a1 | a2, fp32 [*,H] each
  AM_EDGEFEAT     // node_feat[src]|rel_emb[rel]|node_feat[dst]
};

struct GemmP {
  const void* a0; const void* a1; const void* a2;
  const int*  idx; const int* idxr; const int* idxd;
  const float* W;
  const float* bias;      // [N] or nullptr
  const bf16_t* res;      // [M,N] bf16 or nullptr
  float*  outF;           // one of outF/outB non-null
  bf16_t* outB;
  int M, Nn, K, lda;
  int act;                // 0 none, 1 relu, 2 tanh
  int epg;                // edges-per-graph for m/epg indexing
};

template <int AM>
DEV_ float loadA(const GemmP& p, int m, int k) {
  if constexpr (AM == AM_PLAIN)
    return ((const float*)p.a0)[(size_t)m * p.lda + k];
  if constexpr (AM == AM_ADD2)
    return ((const float*)p.a0)[(size_t)m * H_ + k] +
           ((const float*)p.a1)[(size_t)m * H_ + k];
  if constexpr (AM == AM_GATHER)
    return ((const float*)p.a0)[(size_t)p.idx[m] * H_ + k];
  if constexpr (AM == AM_CONCATQ) {
    if (k < H_) return (float)((const bf16_t*)p.a0)[(size_t)m * H_ + k];
    return ((const float*)p.a1)[(size_t)(m / p.epg) * H_ + (k - H_)];
  }
  if constexpr (AM == AM_CONCAT2) {
    if (k < H_) return ((const float*)p.a0)[(size_t)m * H_ + k];
    return ((const float*)p.a1)[(size_t)m * H_ + (k - H_)];
  }
  if constexpr (AM == AM_CONCAT3) {
    if (k < H_)      return ((const float*)p.a0)[(size_t)m * H_ + k];
    if (k < 2 * H_)  return ((const float*)p.a1)[(size_t)m * H_ + (k - H_)];
    return ((const float*)p.a2)[(size_t)m * H_ + (k - 2 * H_)];
  }
  if constexpr (AM == AM_EDGEFEAT) {
    if (k < NEMB_)
      return ((const float*)p.a0)[(size_t)p.idx[m] * NEMB_ + k];
    if (k < NEMB_ + REMB_)
      return ((const float*)p.a1)[(size_t)p.idxr[m] * REMB_ + (k - NEMB_)];
    return ((const float*)p.a0)[(size_t)p.idxd[m] * NEMB_ + (k - NEMB_ - REMB_)];
  }
  return 0.f;
}

// Block tile 128x64, K-step 32, 8 waves (wave32): each wave owns a 32x32
// sub-tile as 2x2 v_wmma_f32_16x16x32_bf16 accumulators.
template <int AM>
__launch_bounds__(256)
__global__ void k_gemm(GemmP p) {
  constexpr int BM = 128, BN = 64, BK = 32;
  constexpr int LDA_ = 40;   // bf16 elems/row (80 B, 16B-aligned)
  constexpr int LDB_ = 80;   // k-major B tile stride (160 B)
  __shared__ __align__(16) bf16_t As[BM * LDA_];
  __shared__ __align__(16) bf16_t Bs[BK * LDB_];

  const int tid   = threadIdx.x;
  const int lane  = tid & 31;
  const int wave  = tid >> 5;
  const int waveM = wave & 3;        // 4 row-groups of 32
  const int waveN = wave >> 2;       // 2 col-groups of 32
  const int half  = lane >> 4;
  const int l16   = lane & 15;
  const int mBase = blockIdx.x * BM;
  const int nBase = blockIdx.y * BN;

  v8f acc[2][2] = {};

  for (int kc = 0; kc < p.K; kc += BK) {
    // Stage A tile (fp32/bf16 virtual source -> bf16 LDS, M-major)
    for (int i = tid; i < BM * BK; i += 256) {
      int r = i >> 5, kk = i & 31;
      As[r * LDA_ + kk] = (bf16_t)loadA<AM>(p, mBase + r, kc + kk);
    }
    // Stage B tile k-major: Bs[k][n] = W[n][k] (fp32 -> bf16)
    for (int i = tid; i < BN * BK; i += 256) {
      int n = i & 63, kk = i >> 6;
      Bs[kk * LDB_ + n] = (bf16_t)p.W[(size_t)(nBase + n) * p.K + (kc + kk)];
    }
    __syncthreads();

    // A frag (ISA 16-bit A 16x32): M=l16, K(j)=16*(j>>3)+8*half+(j&7)
    v16bf afr[2], bfr[2];
#pragma unroll
    for (int mt = 0; mt < 2; ++mt) {
      int row = waveM * 32 + mt * 16 + l16;
      union { uint4 q[2]; v16bf v; } u;
      u.q[0] = *(const uint4*)&As[row * LDA_ + 8 * half];
      u.q[1] = *(const uint4*)&As[row * LDA_ + 16 + 8 * half];
      afr[mt] = u.v;
    }
    // B frag: N = l16, K(j)=16*(j>>3)+8*half+(j&7) (same as A)
    {
#pragma unroll
      for (int nt = 0; nt < 2; ++nt) {
        int colBase = waveN * 32 + nt * 16;
#pragma unroll
        for (int j = 0; j < 16; ++j) {
          int kk = 16 * (j >> 3) + 8 * half + (j & 7);
          bfr[nt][j] = Bs[kk * LDB_ + colBase + l16];
        }
      }
    }
#pragma unroll
    for (int mt = 0; mt < 2; ++mt)
#pragma unroll
      for (int nt = 0; nt < 2; ++nt)
        acc[mt][nt] = __builtin_amdgcn_wmma_f32_16x16x32_bf16(
            false, afr[mt], false, bfr[nt], (short)0, acc[mt][nt], false, false);
    __syncthreads();
  }

  // Epilogue: C layout M = r + 8*half, N = l16
#pragma unroll
  for (int mt = 0; mt < 2; ++mt)
#pragma unroll
    for (int nt = 0; nt < 2; ++nt)
#pragma unroll
      for (int r = 0; r < 8; ++r) {
        int gm = mBase + waveM * 32 + mt * 16 + 8 * half + r;
        int gn = nBase + waveN * 32 + nt * 16 + l16;
        float v = acc[mt][nt][r];
        if (p.bias) v += p.bias[gn];
        size_t off = (size_t)gm * p.Nn + gn;
        if (p.res) v += (float)p.res[off];
        if (p.act == 1) v = fmaxf(v, 0.f);
        else if (p.act == 2) v = tanhf(v);
        if (p.outF) p.outF[off] = v;
        else        p.outB[off] = (bf16_t)v;
      }
}

template <int AM>
static inline void gemm_go(const GemmP& p, hipStream_t st) {
  dim3 grid(p.M / 128, p.Nn / 64);
  k_gemm<AM><<<grid, dim3(256), 0, st>>>(p);
}

// --------------------------- small kernels ---------------------------------

__global__ void k_tgtrel(const float* __restrict__ rel_emb,
                         const int* __restrict__ batch_rel, float* out) {
  int b = blockIdx.x, j = threadIdx.x;
  out[b * H_ + j] = rel_emb[(size_t)batch_rel[b] * REMB_ + j];
}

__global__ void k_gq(const float* __restrict__ input_node,
                     const float* __restrict__ tgt_rel,
                     const int* __restrict__ src, const int* __restrict__ tgt,
                     float* out) {
  int b = blockIdx.x, j = threadIdx.x;
  out[b * H_ + j] = input_node[(size_t)src[b] * H_ + j] + tgt_rel[b * H_ + j]
                  - input_node[(size_t)tgt[b] * H_ + j];
}

__global__ void k_zero(float* p, int n) {
  for (int i = blockIdx.x * blockDim.x + threadIdx.x; i < n;
       i += gridDim.x * blockDim.x)
    p[i] = 0.f;
}

__global__ void k_segsum(const bf16_t* __restrict__ me,
                         const int* __restrict__ dst, float* agg) {
  for (size_t i = (size_t)blockIdx.x * blockDim.x + threadIdx.x;
       i < (size_t)TE_ * H_; i += (size_t)gridDim.x * blockDim.x) {
    int e = (int)(i >> 8), k = (int)(i & 255);
    atomicAdd(&agg[(size_t)dst[e] * H_ + k], (float)me[i]);
  }
}

// per-edge scalar attention: out = in * sigmoid(dot(att[e,:], w2)), wave/edge
__global__ void k_attscale(const bf16_t* __restrict__ att,
                           const float* __restrict__ w2,
                           const bf16_t* __restrict__ in, bf16_t* out) {
  int lane = threadIdx.x & 31;
  int wid  = blockIdx.x * 8 + (threadIdx.x >> 5);
  for (int e = wid; e < TE_; e += gridDim.x * 8) {
    size_t base = (size_t)e * H_;
    float s = 0.f;
#pragma unroll
    for (int i = 0; i < 8; ++i)
      s += (float)att[base + i * 32 + lane] * w2[i * 32 + lane];
#pragma unroll
    for (int off = 16; off; off >>= 1) s += __shfl_xor(s, off, 32);
    s = sigf(s);
#pragma unroll
    for (int i = 0; i < 8; ++i) {
      int k = i * 32 + lane;
      out[base + k] = (bf16_t)((float)in[base + k] * s);
    }
  }
}

__global__ void k_h0(const float* __restrict__ agg2, float* h0) {
  int b = blockIdx.x, j = threadIdx.x;
  float v = agg2[(size_t)(b * NN_) * H_ + j];
  for (int n = 1; n < NN_; ++n)
    v = fmaxf(v, agg2[(size_t)(b * NN_ + n) * H_ + j]);
  h0[b * H_ + j] = v;
}

__global__ void k_msg(float* agg2, const float* __restrict__ gru_bias) {
  for (size_t i = (size_t)blockIdx.x * blockDim.x + threadIdx.x;
       i < (size_t)TN_ * H_; i += (size_t)gridDim.x * blockDim.x)
    agg2[i] = fmaxf(agg2[i] + gru_bias[i & 255], 0.f);
}

__global__ void k_transpose(const float* __restrict__ w, float* wt) {
  // [768,256] -> [256,768]
  for (int i = blockIdx.x * blockDim.x + threadIdx.x; i < 768 * 256;
       i += gridDim.x * blockDim.x) {
    int j = i >> 8, k = i & 255;
    wt[k * 768 + j] = w[i];
  }
}

// Persistent bidirectional GRU: block = (4 graphs, one direction)
__global__ __launch_bounds__(256)
void k_gru(const float* __restrict__ GI_f, const float* __restrict__ GI_b,
           const float* __restrict__ whtF, const float* __restrict__ whtB,
           const float* __restrict__ bhhF, const float* __restrict__ bhhB,
           const float* __restrict__ h0, float* out_f, float* out_b) {
  const int dir = blockIdx.y;
  const float* GI  = dir ? GI_b : GI_f;
  const float* wht = dir ? whtB : whtF;   // transposed [256,768]
  const float* bhh = dir ? bhhB : bhhF;
  float* out = dir ? out_b : out_f;
  const int g0 = blockIdx.x * 4;
  const int j  = threadIdx.x;

  __shared__ float h[4][H_];
#pragma unroll
  for (int g = 0; g < 4; ++g) h[g][j] = h0[(g0 + g) * H_ + j];
  __syncthreads();

  for (int t = 0; t < NN_; ++t) {
    const int tt = dir ? (NN_ - 1 - t) : t;
    float a0[4] = {0, 0, 0, 0}, a1[4] = {0, 0, 0, 0}, a2[4] = {0, 0, 0, 0};
    for (int k = 0; k < H_; ++k) {
      float w0 = wht[k * 768 + j];
      float w1 = wht[k * 768 + 256 + j];
      float w2 = wht[k * 768 + 512 + j];
#pragma unroll
      for (int g = 0; g < 4; ++g) {
        float hv = h[g][k];
        a0[g] = fmaf(w0, hv, a0[g]);
        a1[g] = fmaf(w1, hv, a1[g]);
        a2[g] = fmaf(w2, hv, a2[g]);
      }
    }
    float hn[4];
#pragma unroll
    for (int g = 0; g < 4; ++g) {
      size_t base = ((size_t)(g0 + g) * NN_ + tt) * 768;
      float r = sigf(GI[base + j]       + a0[g] + bhh[j]);
      float z = sigf(GI[base + 256 + j] + a1[g] + bhh[256 + j]);
      float n = tanhf(GI[base + 512 + j] + r * (a2[g] + bhh[512 + j]));
      hn[g] = (1.f - z) * n + z * h[g][j];
    }
    __syncthreads();
#pragma unroll
    for (int g = 0; g < 4; ++g) {
      h[g][j] = hn[g];
      out[((size_t)(g0 + g) * NN_ + tt) * H_ + j] = hn[g];
    }
    __syncthreads();
  }
}

__global__ void k_final(const float* __restrict__ atom_h,
                        const float* __restrict__ tgt_rel,
                        const int* __restrict__ src, const int* __restrict__ tgt,
                        const float* __restrict__ W_lin1,
                        const float* __restrict__ b_lin1, float* out) {
  __shared__ float red[256];
  int b = blockIdx.x, j = threadIdx.x;
  float c = tanhf(atom_h[(size_t)src[b] * H_ + j] + tgt_rel[b * H_ + j]
                - atom_h[(size_t)tgt[b] * H_ + j]);
  red[j] = c * W_lin1[j];
  __syncthreads();
  for (int s = 128; s; s >>= 1) {
    if (j < s) red[j] += red[j + s];
    __syncthreads();
  }
  if (j == 0) out[b] = red[0] + b_lin1[0];
}

// ------------------------------- host -------------------------------------

constexpr size_t SZ_BH   = (size_t)B_ * H_ * 4;      // 128 KiB
constexpr size_t SZ_WT   = (size_t)768 * 256 * 4;    // 768 KiB
constexpr size_t SZ_NODE = (size_t)TN_ * H_ * 4;     // 12.5 MiB
constexpr size_t SZ_EDGE = (size_t)TE_ * H_ * 2;     // 50 MiB (bf16)
constexpr size_t SZ_GI   = (size_t)TN_ * 768 * 4;    // 37.5 MiB

constexpr size_t O_TGTREL = 0;
constexpr size_t O_GQ     = O_TGTREL + SZ_BH;
constexpr size_t O_H0     = O_GQ + SZ_BH;
constexpr size_t O_WHTF   = O_H0 + SZ_BH;
constexpr size_t O_WHTB   = O_WHTF + SZ_WT;
constexpr size_t O_INODE  = O_WHTB + SZ_WT;
constexpr size_t O_NODEA  = O_INODE + SZ_NODE;
constexpr size_t O_NODEB  = O_NODEA + SZ_NODE;
constexpr size_t O_AGG    = O_NODEB + SZ_NODE;
constexpr size_t O_AGG2   = O_AGG + SZ_NODE;
constexpr size_t O_EDGE   = O_AGG2 + SZ_NODE;        // edge-phase region
constexpr size_t O_IEDGE  = O_EDGE;
constexpr size_t O_T2     = O_EDGE + SZ_EDGE;
constexpr size_t O_MEDGE  = O_EDGE + 2 * SZ_EDGE;
// Late-phase buffers alias the (then dead) edge region:
constexpr size_t O_GIF    = O_EDGE;
constexpr size_t O_GIB    = O_EDGE + SZ_GI;
constexpr size_t O_OUTF   = O_EDGE + 2 * SZ_GI;
constexpr size_t O_OUTB   = O_OUTF + SZ_NODE;
constexpr size_t O_ATOMH  = O_OUTB + SZ_NODE;

extern "C" void kernel_launch(void* const* d_in, const int* in_sizes, int n_in,
                              void* d_out, int out_size, void* d_ws,
                              size_t ws_size, hipStream_t stream) {
  (void)in_sizes; (void)n_in; (void)out_size; (void)ws_size;
  const float* node_feat = (const float*)d_in[0];
  const float* rel_emb   = (const float*)d_in[1];
  const float* W_i_node  = (const float*)d_in[2];
  const float* W_i_edge  = (const float*)d_in[3];
  const float* Wa1       = (const float*)d_in[4];
  const float* Wa2       = (const float*)d_in[5];
  const float* W_h_atom  = (const float*)d_in[6];
  const float* W_h_bond  = (const float*)d_in[7];
  const float* Wd1       = (const float*)d_in[8];
  const float* Wd2       = (const float*)d_in[9];
  const float* W_comm    = (const float*)d_in[10];
  const float* gru_bias  = (const float*)d_in[11];
  const float* w_ih_f    = (const float*)d_in[12];
  const float* w_hh_f    = (const float*)d_in[13];
  const float* b_ih_f    = (const float*)d_in[14];
  const float* b_hh_f    = (const float*)d_in[15];
  const float* w_ih_b    = (const float*)d_in[16];
  const float* w_hh_b    = (const float*)d_in[17];
  const float* b_ih_b    = (const float*)d_in[18];
  const float* b_hh_b    = (const float*)d_in[19];
  const float* W_o       = (const float*)d_in[20];
  const float* b_o       = (const float*)d_in[21];
  const float* W_lin1    = (const float*)d_in[22];
  const float* b_lin1    = (const float*)d_in[23];
  const int* edge_src    = (const int*)d_in[24];
  const int* edge_dst    = (const int*)d_in[25];
  const int* edge_rel    = (const int*)d_in[26];
  const int* batch_rel   = (const int*)d_in[27];
  const int* src_node    = (const int*)d_in[28];
  const int* tgt_node    = (const int*)d_in[29];

  char* ws = (char*)d_ws;
  float*  tgt_rel    = (float*)(ws + O_TGTREL);
  float*  gq         = (float*)(ws + O_GQ);
  float*  h0         = (float*)(ws + O_H0);
  float*  whtF       = (float*)(ws + O_WHTF);
  float*  whtB       = (float*)(ws + O_WHTB);
  float*  input_node = (float*)(ws + O_INODE);
  float*  nodeA      = (float*)(ws + O_NODEA);
  float*  nodeB      = (float*)(ws + O_NODEB);
  float*  agg        = (float*)(ws + O_AGG);
  float*  agg2       = (float*)(ws + O_AGG2);
  bf16_t* input_edge = (bf16_t*)(ws + O_IEDGE);
  bf16_t* T2         = (bf16_t*)(ws + O_T2);
  bf16_t* mess_edge  = (bf16_t*)(ws + O_MEDGE);
  float*  GI_f       = (float*)(ws + O_GIF);
  float*  GI_b       = (float*)(ws + O_GIB);
  float*  out_f      = (float*)(ws + O_OUTF);
  float*  out_b      = (float*)(ws + O_OUTB);
  float*  atom_h     = (float*)(ws + O_ATOMH);
  float*  out        = (float*)d_out;

  // 0) per-graph query relation embedding + w_hh transposes
  k_tgtrel<<<B_, H_, 0, stream>>>(rel_emb, batch_rel, tgt_rel);
  k_transpose<<<512, 256, 0, stream>>>(w_hh_f, whtF);
  k_transpose<<<512, 256, 0, stream>>>(w_hh_b, whtB);

  GemmP p;

  // 1) input_node = relu(node_feat @ W_i_node^T)
  p = {}; p.a0 = node_feat; p.W = W_i_node; p.outF = input_node;
  p.M = TN_; p.Nn = H_; p.K = NEMB_; p.lda = NEMB_; p.act = 1; p.epg = 1;
  gemm_go<AM_PLAIN>(p, stream);

  // 2) per-graph translational query: gq = in[src] + tgt_rel - in[tgt]
  k_gq<<<B_, H_, 0, stream>>>(input_node, tgt_rel, src_node, tgt_node, gq);

  // 3) input_edge = relu([nf[src]|rel|nf[dst]] @ W_i_edge^T)   (bf16 out)
  p = {}; p.a0 = node_feat; p.a1 = rel_emb;
  p.idx = edge_src; p.idxr = edge_rel; p.idxd = edge_dst;
  p.W = W_i_edge; p.outB = input_edge;
  p.M = TE_; p.Nn = H_; p.K = 2 * NEMB_ + REMB_; p.act = 1; p.epg = 1;
  gemm_go<AM_EDGEFEAT>(p, stream);

  // 4) T2 = tanh(concat(input_edge, gq) @ Wa1^T)
  p = {}; p.a0 = input_edge; p.a1 = gq; p.W = Wa1; p.outB = T2;
  p.M = TE_; p.Nn = H_; p.K = 2 * H_; p.act = 2; p.epg = EE_;
  gemm_go<AM_CONCATQ>(p, stream);

  // 5) message_edge = input_edge * sigmoid(T2 @ Wa2^T)
  k_attscale<<<2048, 256, 0, stream>>>(T2, Wa2, input_edge, mess_edge);

  // 6) depth loop
  const float* mn_prev = input_node;
  float* node_buf[2] = {nodeA, nodeB};
  for (int d = 0; d < 2; ++d) {
    k_zero<<<2048, 256, 0, stream>>>(agg, TN_ * H_);
    k_segsum<<<4096, 256, 0, stream>>>(mess_edge, edge_dst, agg);

    // message_node = relu((mn_prev + agg) @ W_h_atom[d]^T)
    p = {}; p.a0 = mn_prev; p.a1 = agg;
    p.W = W_h_atom + (size_t)d * H_ * H_; p.outF = node_buf[d];
    p.M = TN_; p.Nn = H_; p.K = H_; p.act = 1; p.epg = 1;
    gemm_go<AM_ADD2>(p, stream);
    mn_prev = node_buf[d];

    // message_edge = relu(input_edge + mn[edge_src] @ W_h_bond[d]^T)
    p = {}; p.a0 = node_buf[d]; p.idx = edge_src;
    p.W = W_h_bond + (size_t)d * H_ * H_; p.res = input_edge;
    p.outB = mess_edge;
    p.M = TE_; p.Nn = H_; p.K = H_; p.act = 1; p.epg = 1;
    gemm_go<AM_GATHER>(p, stream);

    // T2 = tanh(concat(message_edge, tgt_rel) @ Wd1[d]^T)
    p = {}; p.a0 = mess_edge; p.a1 = tgt_rel;
    p.W = Wd1 + (size_t)d * H_ * 2 * H_; p.outB = T2;
    p.M = TE_; p.Nn = H_; p.K = 2 * H_; p.act = 2; p.epg = EE_;
    gemm_go<AM_CONCATQ>(p, stream);

    // message_edge *= sigmoid(T2 @ Wd2[d]^T)
    k_attscale<<<2048, 256, 0, stream>>>(T2, Wd2 + (size_t)d * H_,
                                         mess_edge, mess_edge);
  }

  // 7) final aggregation
  k_zero<<<2048, 256, 0, stream>>>(agg, TN_ * H_);
  k_segsum<<<4096, 256, 0, stream>>>(mess_edge, edge_dst, agg);

  // 8) agg2 = concat(agg, message_node, input_node) @ W_comm^T
  p = {}; p.a0 = agg; p.a1 = nodeB; p.a2 = input_node;
  p.W = W_comm; p.outF = agg2;
  p.M = TN_; p.Nn = H_; p.K = 3 * H_; p.act = 0; p.epg = 1;
  gemm_go<AM_CONCAT3>(p, stream);

  // 9) h0 = per-graph max(agg2);  msg = relu(agg2 + gru_bias)  (in place)
  k_h0<<<B_, H_, 0, stream>>>(agg2, h0);
  k_msg<<<2048, 256, 0, stream>>>(agg2, gru_bias);

  // 10) GRU input projections (edge-phase buffers are dead; GI aliases them)
  p = {}; p.a0 = agg2; p.W = w_ih_f; p.bias = b_ih_f; p.outF = GI_f;
  p.M = TN_; p.Nn = 768; p.K = H_; p.lda = H_; p.act = 0; p.epg = 1;
  gemm_go<AM_PLAIN>(p, stream);
  p.W = w_ih_b; p.bias = b_ih_b; p.outF = GI_b;
  gemm_go<AM_PLAIN>(p, stream);

  // 11) bidirectional GRU recurrence
  k_gru<<<dim3(B_ / 4, 2), 256, 0, stream>>>(GI_f, GI_b, whtF, whtB,
                                             b_hh_f, b_hh_b, h0, out_f, out_b);

  // 12) atom_h = relu(concat(out_f, out_b) @ W_o^T + b_o)
  p = {}; p.a0 = out_f; p.a1 = out_b; p.W = W_o; p.bias = b_o; p.outF = atom_h;
  p.M = TN_; p.Nn = H_; p.K = 2 * H_; p.act = 1; p.epg = 1;
  gemm_go<AM_CONCAT2>(p, stream);

  // 13) out = tanh(atom_h[src] + tgt_rel - atom_h[tgt]) @ W_lin1^T + b_lin1
  k_final<<<B_, 256, 0, stream>>>(atom_h, tgt_rel, src_node, tgt_node,
                                  W_lin1, b_lin1, out);
}
